// EmotionOV_14035953123713
// MI455X (gfx1250) — matrix operand order repaired and checked by reading, not compiled, not run on hardware
//
#include <hip/hip_runtime.h>
#include <math.h>

// ---------------------------------------------------------------------------
// Dims (compile-time constants from the reference)
// ---------------------------------------------------------------------------
#define T_  64
#define S_  729
#define H_  1152
#define E_  512
#define NE_ 9
#define NL_ 4
#define B_  8
#define P_  8

#define KSTEPS   (H_ / 32)          // 36
#define BTILE    (32 * 512)         // bf16 elements per K-step weight tile
#define A_STRIDE 40                 // padded bf16 row stride for A tile

typedef __attribute__((ext_vector_type(16))) __bf16 v16bf;
typedef __attribute__((ext_vector_type(8)))  __bf16 v8bf;
typedef __attribute__((ext_vector_type(4)))  __bf16 v4bf;
typedef __attribute__((ext_vector_type(2)))  __bf16 v2bf;
typedef __attribute__((ext_vector_type(8)))  float  v8f;

// ---------------------------------------------------------------------------
// WMMA fragment helpers (gfx1250 wave32; A/B in this code's contiguous k order, D per ISA 7.12.2)
// A: 16(M) x 32(K) bf16, row stride A_STRIDE. lane L: row=L&15,
//    a[0..15] = K run starting at 16*(L>>4).
// B: swizzled [n_tile][lane][16]: lane L holds B[k=e+16*(L>>4)][n=L&15].
// D: 8 x f32, lane L: col=L&15, vgpr v -> row v + 8*(L>=16).
// ---------------------------------------------------------------------------
__device__ inline v16bf frag_cat(v8bf lo, v8bf hi) {
  return __builtin_shufflevector(lo, hi, 0, 1, 2, 3, 4, 5, 6, 7,
                                 8, 9, 10, 11, 12, 13, 14, 15);
}

__device__ inline v16bf load_frag_A(const __bf16* A, int lane) {
  int row = lane & 15, hi = lane >> 4;
  v8bf lo = *(const v8bf*)(A + row * A_STRIDE + hi * 16);
  v8bf hv = *(const v8bf*)(A + row * A_STRIDE + hi * 16 + 8);
  return frag_cat(lo, hv);
}

__device__ inline v16bf load_frag_B(const __bf16* Bsw, int nt, int lane) {
  const __bf16* p = Bsw + ((nt * 32 + lane) << 4);
  v8bf lo = *(const v8bf*)(p);
  v8bf hv = *(const v8bf*)(p + 8);
  return frag_cat(lo, hv);
}

// Shared tile memory reused between GEMM staging and the f32 result tile.
union __align__(16) SmemU {
  struct { __bf16 A[16 * A_STRIDE]; __bf16 B[BTILE]; } g;  // 1280B + 32KB
  float C[16][520];                                        // 33.3KB
};

// Stage one 16x32 A slice (fp32 global -> bf16 LDS), branchless, b32 store.
__device__ inline void stage_A(__bf16* Adst, const float* Ag, int s0, int k0,
                               int tid) {
  int idx2 = tid * 2;                 // 512 elements total
  int row = idx2 >> 5, kk = idx2 & 31;
  int s = s0 + row;
  int sc = (s < S_) ? s : (S_ - 1);
  float msk = (s < S_) ? 1.f : 0.f;
  const float* src = Ag + (size_t)sc * H_ + k0 + kk;
  v2bf pk;
  pk[0] = (__bf16)(src[0] * msk);
  pk[1] = (__bf16)(src[1] * msk);
  *(v2bf*)(Adst + row * A_STRIDE + kk) = pk;
}

// Async copy of one prepacked 32KB weight tile (global -> LDS), wave32.
// GLOBAL_LOAD_ASYNC_TO_LDS_B128: each lane moves 16B; tracked by ASYNCcnt.
// Generic LDS pointers carry the LDS byte offset in addr[31:0] (ISA 10.2).
__device__ inline void async_copy_btile(__bf16* ldst, const __bf16* gsrc,
                                        int tid) {
  unsigned lds = (unsigned)(size_t)ldst + tid * 16;
  const char* g = (const char*)gsrc + tid * 16;
#pragma unroll
  for (int i = 0; i < 8; ++i) {  // 256 thr * 8 * 16B = 32KB
    asm volatile("global_load_async_to_lds_b128 %0, %1, off offset:%2"
                 :: "v"(lds), "v"(g), "i"(i * 4096) : "memory");
  }
}

__device__ inline void wait_async() {
  asm volatile("s_wait_asynccnt 0x0" ::: "memory");
}

__device__ inline float gelu_exact(float x) {
  return 0.5f * x * (1.0f + erff(x * 0.7071067811865475f));
}

__device__ inline float block_reduce_max(float v, float* red, int tid) {
  red[tid] = v; __syncthreads();
  for (int off = 128; off > 0; off >>= 1) {
    if (tid < off) red[tid] = fmaxf(red[tid], red[tid + off]);
    __syncthreads();
  }
  float r = red[0]; __syncthreads();
  return r;
}

__device__ inline float block_reduce_sum(float v, float* red, int tid) {
  red[tid] = v; __syncthreads();
  for (int off = 128; off > 0; off >>= 1) {
    if (tid < off) red[tid] += red[tid + off];
    __syncthreads();
  }
  float r = red[0]; __syncthreads();
  return r;
}

// ---------------------------------------------------------------------------
// K0: pre-pack weights (sig_w[0..3], k_w) to bf16 in swizzled fragment order.
// grid: (KSTEPS, 5). Output tile (id, kt) is one contiguous 32KB block.
// ---------------------------------------------------------------------------
__global__ __launch_bounds__(256) void pack_w_kernel(
    const float* __restrict__ sig_w, const float* __restrict__ k_w,
    __bf16* __restrict__ Bpk) {
  int kt = blockIdx.x, id = blockIdx.y, tid = threadIdx.x;
  const float* W = (id < NL_) ? (sig_w + (size_t)id * 512 * H_) : k_w;
  __bf16* dst = Bpk + ((size_t)id * KSTEPS + kt) * BTILE;
  int k0 = kt * 32;
#pragma unroll 4
  for (int r = 0; r < 16; ++r) {
    int idx4 = r * 1024 + tid * 4;    // 16384 elements total
    int o = idx4 >> 5, kk = idx4 & 31;
    const float4 v = *(const float4*)(W + (size_t)o * H_ + k0 + kk);
    int di = (((o >> 4) * 32 + ((kk >> 4) << 4) + (o & 15)) << 4) + (kk & 15);
    v4bf pk;
    pk[0] = (__bf16)v.x; pk[1] = (__bf16)v.y;
    pk[2] = (__bf16)v.z; pk[3] = (__bf16)v.w;
    *(v4bf*)(dst + di) = pk;
  }
}

// ---------------------------------------------------------------------------
// K1: pooled[b][h] = sum_s ov[b*P, s, h] * pool_w[s] + pool_b
// ---------------------------------------------------------------------------
__global__ __launch_bounds__(256) void pooled_kernel(
    const float* __restrict__ ov, const float* __restrict__ pool_w,
    const float* __restrict__ pool_b, float* __restrict__ pooled) {
  int b = blockIdx.x, tid = threadIdx.x;
  const float* base = ov + (size_t)b * P_ * S_ * H_;  // first patch of image b
  for (int h = tid; h < H_; h += 256) {
    float s = pool_b[0];
    for (int sp = 0; sp < S_; ++sp) s += base[(size_t)sp * H_ + h] * pool_w[sp];
    pooled[b * H_ + h] = s;
  }
}

// ---------------------------------------------------------------------------
// K2: cls[b] = gelu(pooled @ W1^T + b1) @ W2^T + b2   (per image)
// ---------------------------------------------------------------------------
__global__ __launch_bounds__(256) void cls_kernel(
    const float* __restrict__ pooled, const float* __restrict__ w1,
    const float* __restrict__ b1, const float* __restrict__ w2,
    const float* __restrict__ b2, float* __restrict__ cls) {
  int b = blockIdx.x, tid = threadIdx.x;
  __shared__ float pl[H_];
  __shared__ float hid[512];
  for (int h = tid; h < H_; h += 256) pl[h] = pooled[b * H_ + h];
  __syncthreads();
  for (int c = tid; c < 512; c += 256) {
    float s = b1[c];
    const float* w = w1 + (size_t)c * H_;
    for (int h = 0; h < H_; ++h) s += pl[h] * w[h];
    hid[c] = gelu_exact(s);
  }
  __syncthreads();
  for (int c = tid; c < 512; c += 256) {
    float s = b2[c];
    const float* w = w2 + (size_t)c * 512;
    for (int x = 0; x < 512; ++x) s += hid[x] * w[x];
    cls[b * 512 + c] = s;
  }
}

// ---------------------------------------------------------------------------
// K3: prompts[i][b][n][:] = l2norm(cat(emo[n], cls[b]) @ prompt_w[i]^T + pb[i])
// ---------------------------------------------------------------------------
__global__ __launch_bounds__(256) void prompt_kernel(
    const float* __restrict__ emo, const float* __restrict__ cls,
    const float* __restrict__ pw, const float* __restrict__ pb,
    float* __restrict__ prompts) {
  int i = blockIdx.x / 72, rem = blockIdx.x % 72;
  int b = rem / 9, n = rem % 9;
  int tid = threadIdx.x;
  __shared__ float cat[1024];
  __shared__ float raw[512];
  __shared__ float red[256];
  for (int c = tid; c < 512; c += 256) {
    cat[c] = emo[n * 512 + c];
    cat[512 + c] = cls[b * 512 + c];
  }
  __syncthreads();
  for (int o = tid; o < 512; o += 256) {
    float s = pb[i * 512 + o];
    const float* w = pw + ((size_t)i * 512 + o) * 1024;
    for (int c = 0; c < 1024; ++c) s += cat[c] * w[c];
    raw[o] = s;
  }
  __syncthreads();
  float sq = 0.f;
  for (int o = tid; o < 512; o += 256) sq += raw[o] * raw[o];
  float nrm2 = block_reduce_sum(sq, red, tid);
  float rn = rsqrtf(nrm2);
  float* dst = prompts + (((size_t)i * B_ + b) * NE_ + n) * 512;
  for (int o = tid; o < 512; o += 256) dst[o] = raw[o] * rn;
}

// ---------------------------------------------------------------------------
// K4 (WMMA): fused sig_emb GEMM + L2-norm + prompt contraction.
// Per block: (level i, patch t, 16-row s-tile).
//   C[16x512] = sig[i,t,s0:s0+16,:] @ sig_w[i]^T + sig_b[i]     (bf16 WMMA)
//   rows L2-normalized, then scores[i,t,n,s] = 100 * C_n @ prompts[i,t,n]
// Weights come prepacked; staging uses async global->LDS b128 copies.
// ---------------------------------------------------------------------------
__global__ __launch_bounds__(256) void sig_scores_kernel(
    const float* __restrict__ sig, const __bf16* __restrict__ Bpk,
    const float* __restrict__ sig_b, const float* __restrict__ prompts,
    float* __restrict__ scores) {
  int mb = blockIdx.x, t = blockIdx.y, lvl = blockIdx.z;
  int s0 = mb * 16;
  int tid = threadIdx.x, lane = tid & 31, wave = tid >> 5;

  __shared__ SmemU sm;
  __shared__ float part[256];
  __shared__ float rn[16];

  const float* Ag = sig + ((size_t)lvl * T_ + t) * S_ * H_;
  const __bf16* Wp = Bpk + (size_t)lvl * KSTEPS * BTILE;

  v8f acc[4];
#pragma unroll
  for (int j = 0; j < 4; ++j)
#pragma unroll
    for (int e = 0; e < 8; ++e) acc[j][e] = 0.f;

  for (int kt = 0; kt < KSTEPS; ++kt) {
    async_copy_btile(sm.g.B, Wp + (size_t)kt * BTILE, tid);
    stage_A(sm.g.A, Ag, s0, kt * 32, tid);
    if (kt + 1 < KSTEPS)  // prefetch next prepacked tile (global_prefetch_b8)
      __builtin_prefetch((const char*)(Wp + (size_t)(kt + 1) * BTILE) + tid * 128, 0, 1);
    wait_async();
    __syncthreads();
    v16bf af = load_frag_A(sm.g.A, lane);
#pragma unroll
    for (int j = 0; j < 4; ++j) {
      v16bf bf = load_frag_B(sm.g.B, wave * 4 + j, lane);
      acc[j] = __builtin_amdgcn_wmma_f32_16x16x32_bf16(
          false, af, false, bf, (short)0, acc[j], false, false);
    }
    __syncthreads();
  }

  // store accumulators (+bias) into f32 tile (reuses staging space)
  {
    int hi = lane >> 4, col = lane & 15;
#pragma unroll
    for (int j = 0; j < 4; ++j) {
      int n0 = (wave * 4 + j) * 16;
#pragma unroll
      for (int v = 0; v < 8; ++v) {
        int row = v + hi * 8;
        sm.C[row][n0 + col] = acc[j][v] + sig_b[lvl * 512 + n0 + col];
      }
    }
  }
  __syncthreads();

  // row L2 norms: 16 threads per row, 32 elements each
  {
    int row = tid >> 4, c0 = (tid & 15) * 32;
    float sq = 0.f;
    for (int c = 0; c < 32; ++c) { float v = sm.C[row][c0 + c]; sq += v * v; }
    part[tid] = sq;
    __syncthreads();
    if (tid < 16) {
      float s = 0.f;
      for (int j = 0; j < 16; ++j) s += part[tid * 16 + j];
      rn[tid] = rsqrtf(s);
    }
    __syncthreads();
  }

  // scores against 9 prompts (prompts depend only on image b = t/8)
  const float* Pr = prompts + (((size_t)lvl * B_ + (t >> 3)) * NE_) * 512;
  if (tid < 16 * NE_) {
    int row = tid / NE_, ne = tid % NE_;
    int s = s0 + row;
    if (s < S_) {
      const float* pv = Pr + ne * 512;
      float sum = 0.f;
      for (int o = 0; o < 512; ++o) sum += sm.C[row][o] * pv[o];
      scores[(((size_t)lvl * T_ + t) * NE_ + ne) * S_ + s] = 100.f * sum * rn[row];
    }
  }
}

// ---------------------------------------------------------------------------
// K5: per (t, ne): softmax over S for each level, mean over 4 levels,
// plus patch_emotion_scores = mean over (level, s).
// ---------------------------------------------------------------------------
__global__ __launch_bounds__(256) void softmax_kernel(
    const float* __restrict__ scores, float* __restrict__ attm,
    float* __restrict__ pscores) {
  int ne = blockIdx.x, t = blockIdx.y, tid = threadIdx.x;
  __shared__ float red[256];
  float a0 = 0.f, a1 = 0.f, a2 = 0.f, ps = 0.f;
  for (int lvl = 0; lvl < NL_; ++lvl) {
    const float* row = scores + (((size_t)lvl * T_ + t) * NE_ + ne) * S_;
    float x0 = (tid < S_) ? row[tid] : -3.0e38f;
    float x1 = (tid + 256 < S_) ? row[tid + 256] : -3.0e38f;
    float x2 = (tid + 512 < S_) ? row[tid + 512] : -3.0e38f;
    float m = block_reduce_max(fmaxf(x0, fmaxf(x1, x2)), red, tid);
    float e0 = (tid < S_) ? __expf(x0 - m) : 0.f;
    float e1 = (tid + 256 < S_) ? __expf(x1 - m) : 0.f;
    float e2 = (tid + 512 < S_) ? __expf(x2 - m) : 0.f;
    float tot = block_reduce_sum(e0 + e1 + e2, red, tid);
    float inv = 1.f / tot;
    a0 += 0.25f * e0 * inv;
    a1 += 0.25f * e1 * inv;
    a2 += 0.25f * e2 * inv;
    ps += (e0 + e1 + e2) * inv;
  }
  float* dst = attm + ((size_t)t * NE_ + ne) * S_;
  if (tid < S_) dst[tid] = a0;
  if (tid + 256 < S_) dst[tid + 256] = a1;
  if (tid + 512 < S_) dst[tid + 512] = a2;
  float tot = block_reduce_sum(ps, red, tid);
  if (tid == 0) pscores[t * NE_ + ne] = tot / (float)(NL_ * S_);
}

// ---------------------------------------------------------------------------
// K6: per t: max_att over emotions, overlapping 2x2 pooling (bins 0:14/13:27),
// tokens[t][q][h] and global_sig[t][q].
// ---------------------------------------------------------------------------
__global__ __launch_bounds__(256) void tokens_kernel(
    const float* __restrict__ attm, const float* __restrict__ pscores,
    const float* __restrict__ ov, float* __restrict__ tokens,
    float* __restrict__ gsig) {
  int t = blockIdx.x, tid = threadIdx.x;
  __shared__ float ma[S_];
  __shared__ unsigned char bflag[S_];
  __shared__ float mb[4];
  __shared__ float psx;
  if (tid < 4) mb[tid] = 0.f;
  if (tid == 0) {
    float m = pscores[t * NE_];
    for (int ne = 1; ne < NE_; ++ne) m = fmaxf(m, pscores[t * NE_ + ne]);
    psx = m;
  }
  for (int s = tid; s < S_; s += 256) {
    float m = attm[((size_t)t * NE_) * S_ + s];
    for (int ne = 1; ne < NE_; ++ne)
      m = fmaxf(m, attm[((size_t)t * NE_ + ne) * S_ + s]);
    ma[s] = m;
    int r = s / 27, c = s % 27;
    unsigned char f = 0;
    if (r <= 13 && c <= 13) f |= 1;
    if (r <= 13 && c >= 13) f |= 2;
    if (r >= 13 && c <= 13) f |= 4;
    if (r >= 13 && c >= 13) f |= 8;
    bflag[s] = f;
  }
  __syncthreads();
  for (int s = tid; s < S_; s += 256) {
    float v = ma[s];
    unsigned char f = bflag[s];
    if (f & 1) atomicAdd(&mb[0], v);
    if (f & 2) atomicAdd(&mb[1], v);
    if (f & 4) atomicAdd(&mb[2], v);
    if (f & 8) atomicAdd(&mb[3], v);
  }
  __syncthreads();
  float inv196 = 1.f / 196.f;
  float dq[4];
#pragma unroll
  for (int q = 0; q < 4; ++q) dq[q] = 1.f / (mb[q] * inv196 + 1e-8f);
  if (tid < 4) gsig[t * 4 + tid] = (mb[tid] * inv196) * psx;
  const float* ovt = ov + (size_t)t * S_ * H_;
  for (int hc = 0; hc < 5; ++hc) {
    int h = hc * 256 + tid;
    if (h >= H_) break;
    float a0 = 0.f, a1 = 0.f, a2 = 0.f, a3 = 0.f;
    for (int s = 0; s < S_; ++s) {
      float v = ovt[(size_t)s * H_ + h] * ma[s];
      unsigned char f = bflag[s];
      if (f & 1) a0 += v;
      if (f & 2) a1 += v;
      if (f & 4) a2 += v;
      if (f & 8) a3 += v;
    }
    float* dst = tokens + (size_t)t * 4 * H_;
    dst[0 * H_ + h] = a0 * inv196 * dq[0];
    dst[1 * H_ + h] = a1 * inv196 * dq[1];
    dst[2 * H_ + h] = a2 * inv196 * dq[2];
    dst[3 * H_ + h] = a3 * inv196 * dq[3];
  }
}

// ---------------------------------------------------------------------------
// K7 (WMMA): fused q-former attention per patch t (flash-style).
// ---------------------------------------------------------------------------
__global__ __launch_bounds__(256) void attn_kernel(
    const float* __restrict__ ov, const float* __restrict__ tokens,
    const float* __restrict__ qw, const float* __restrict__ qb,
    const __bf16* __restrict__ Kpk, const float* __restrict__ kb,
    float* __restrict__ out_major) {
  int t = blockIdx.x, tid = threadIdx.x, lane = tid & 31, wave = tid >> 5;
  __shared__ float Qs[4][512];
  __shared__ float accV[4][H_];
  __shared__ SmemU sm;  // GEMM staging, then f32 K-tile [16][520]
  __shared__ float lg[4][16], pp[4][16], mrow[4], lrow[4], resc[4];

  const float* ovt = ov + (size_t)t * S_ * H_;
  const float* tok = tokens + (size_t)t * 4 * H_;

  // Q projection
  for (int r = 0; r < 8; ++r) {
    int idx = r * 256 + tid;
    int q = idx >> 9, e = idx & 511;
    float s = qb[e];
    const float* w = qw + (size_t)e * H_;
    for (int h = 0; h < H_; ++h) s += tok[q * H_ + h] * w[h];
    Qs[q][e] = s;
  }
  for (int idx = tid; idx < 4 * H_; idx += 256) (&accV[0][0])[idx] = 0.f;
  if (tid < 4) { mrow[tid] = -3.0e38f; lrow[tid] = 0.f; }
  __syncthreads();

  const float scale = 0.044194173824159216f;  // 1/sqrt(512)
  for (int st = 0; st < 46; ++st) {
    int s0 = st * 16;
    v8f acc[4];
#pragma unroll
    for (int j = 0; j < 4; ++j)
#pragma unroll
      for (int e = 0; e < 8; ++e) acc[j][e] = 0.f;

    for (int kt = 0; kt < KSTEPS; ++kt) {
      async_copy_btile(sm.g.B, Kpk + (size_t)kt * BTILE, tid);
      stage_A(sm.g.A, ovt, s0, kt * 32, tid);
      if (kt + 1 < KSTEPS)
        __builtin_prefetch((const char*)(Kpk + (size_t)(kt + 1) * BTILE) + tid * 128, 0, 1);
      wait_async();
      __syncthreads();
      v16bf af = load_frag_A(sm.g.A, lane);
#pragma unroll
      for (int j = 0; j < 4; ++j) {
        v16bf bf = load_frag_B(sm.g.B, wave * 4 + j, lane);
        acc[j] = __builtin_amdgcn_wmma_f32_16x16x32_bf16(
            false, af, false, bf, (short)0, acc[j], false, false);
      }
      __syncthreads();
    }
    // K tile (+bias) into f32 LDS
    {
      int hi = lane >> 4, col = lane & 15;
#pragma unroll
      for (int j = 0; j < 4; ++j) {
        int n0 = (wave * 4 + j) * 16;
#pragma unroll
        for (int v = 0; v < 8; ++v) {
          int row = v + hi * 8;
          sm.C[row][n0 + col] = acc[j][v] + kb[n0 + col];
        }
      }
    }
    __syncthreads();
    // logits: 4 queries x 16 rows
    if (tid < 64) {
      int q = tid >> 4, r = tid & 15;
      int s = s0 + r;
      float sum = -3.0e38f;
      if (s < S_) {
        sum = 0.f;
        for (int e = 0; e < 512; ++e) sum += Qs[q][e] * sm.C[r][e];
        sum *= scale;
      }
      lg[q][r] = sum;
    }
    __syncthreads();
    // online softmax state
    if (tid < 4) {
      int q = tid;
      float tmax = lg[q][0];
      for (int r = 1; r < 16; ++r) tmax = fmaxf(tmax, lg[q][r]);
      float newm = fmaxf(mrow[q], tmax);
      float rs = __expf(mrow[q] - newm);
      float psum = 0.f;
      for (int r = 0; r < 16; ++r) {
        float p = __expf(lg[q][r] - newm);
        pp[q][r] = p;
        psum += p;
      }
      lrow[q] = lrow[q] * rs + psum;
      mrow[q] = newm;
      resc[q] = rs;
    }
    __syncthreads();
    // accumulate p @ V  (V = raw ov rows)
    for (int hc = 0; hc < 5; ++hc) {
      int h = hc * 256 + tid;
      if (h >= H_) break;
      float s0a = 0.f, s1a = 0.f, s2a = 0.f, s3a = 0.f;
      for (int r = 0; r < 16; ++r) {
        int s = s0 + r;
        if (s >= S_) break;
        float v = ovt[(size_t)s * H_ + h];
        s0a += pp[0][r] * v; s1a += pp[1][r] * v;
        s2a += pp[2][r] * v; s3a += pp[3][r] * v;
      }
      accV[0][h] = accV[0][h] * resc[0] + s0a;
      accV[1][h] = accV[1][h] * resc[1] + s1a;
      accV[2][h] = accV[2][h] * resc[2] + s2a;
      accV[3][h] = accV[3][h] * resc[3] + s3a;
    }
    __syncthreads();
  }
  for (int idx = tid; idx < 4 * H_; idx += 256) {
    int q = idx / H_, h = idx % H_;
    out_major[((size_t)t * 4 + q) * H_ + h] = accV[q][h] / lrow[q];
  }
}

// ---------------------------------------------------------------------------
// K8: per image: significance-weighted mean of major, prediction MLP + softmax
// ---------------------------------------------------------------------------
__global__ __launch_bounds__(256) void final_kernel(
    const float* __restrict__ major, const float* __restrict__ gsig,
    const float* __restrict__ pw1, const float* __restrict__ pb1,
    const float* __restrict__ pw2, const float* __restrict__ pb2,
    float* __restrict__ out_final, float* __restrict__ out_preds) {
  int b = blockIdx.x, tid = threadIdx.x;
  __shared__ float gsh[32];
  __shared__ float fin[H_];
  __shared__ float hid[512];
  __shared__ float red[16];
  if (tid < 32) gsh[tid] = gsig[(b * P_ + tid / 4) * 4 + (tid & 3)];
  __syncthreads();
  if (tid == 0) {
    float s = 0.f;
    for (int j = 0; j < 32; ++j) s += gsh[j];
    red[0] = s;
  }
  __syncthreads();
  float den = red[0] / 32.f + 1e-8f;
  for (int h = tid; h < H_; h += 256) {
    float s = 0.f;
    for (int j = 0; j < 32; ++j) {
      int tt = b * P_ + j / 4, q = j & 3;
      s += major[((size_t)tt * 4 + q) * H_ + h] * gsh[j];
    }
    float f = (s / 32.f) / den;
    fin[h] = f;
    out_final[b * H_ + h] = f;
  }
  __syncthreads();
  for (int c = tid; c < 512; c += 256) {
    float s = pb1[c];
    const float* w = pw1 + (size_t)c * H_;
    for (int h = 0; h < H_; ++h) s += fin[h] * w[h];
    hid[c] = gelu_exact(s);
  }
  __syncthreads();
  if (tid < NE_) {
    float s = pb2[tid];
    const float* w = pw2 + (size_t)tid * 512;
    for (int c = 0; c < 512; ++c) s += hid[c] * w[c];
    red[tid] = s;
  }
  __syncthreads();
  if (tid == 0) {
    float m = red[0];
    for (int n = 1; n < NE_; ++n) m = fmaxf(m, red[n]);
    float tot = 0.f, e[NE_];
    for (int n = 0; n < NE_; ++n) { e[n] = __expf(red[n] - m); tot += e[n]; }
    for (int n = 0; n < NE_; ++n) out_preds[b * NE_ + n] = e[n] / tot;
  }
}

// ---------------------------------------------------------------------------
// Launch
// ---------------------------------------------------------------------------
extern "C" void kernel_launch(void* const* d_in, const int* in_sizes, int n_in,
                              void* d_out, int out_size, void* d_ws, size_t ws_size,
                              hipStream_t stream) {
  const float* ov       = (const float*)d_in[0];
  const float* sig      = (const float*)d_in[1];
  const float* emo      = (const float*)d_in[2];
  const float* pool_w   = (const float*)d_in[3];
  const float* pool_b   = (const float*)d_in[4];
  const float* cls_w1   = (const float*)d_in[5];
  const float* cls_b1   = (const float*)d_in[6];
  const float* cls_w2   = (const float*)d_in[7];
  const float* cls_b2   = (const float*)d_in[8];
  const float* sig_w    = (const float*)d_in[9];
  const float* sig_b    = (const float*)d_in[10];
  const float* prompt_w = (const float*)d_in[11];
  const float* prompt_b = (const float*)d_in[12];
  const float* pred_w1  = (const float*)d_in[13];
  const float* pred_b1  = (const float*)d_in[14];
  const float* pred_w2  = (const float*)d_in[15];
  const float* pred_b2  = (const float*)d_in[16];
  const float* q_w      = (const float*)d_in[17];
  const float* q_b      = (const float*)d_in[18];
  const float* k_w      = (const float*)d_in[19];
  const float* k_b      = (const float*)d_in[20];

  float* ws = (float*)d_ws;
  float* pooled  = ws;                 // 8*1152      = 9216
  float* cls     = ws + 9216;          // 8*512       = 4096
  float* prompts = ws + 13312;         // 4*8*9*512   = 147456
  float* scores  = ws + 160768;        // 4*64*9*729  = 1679616
  float* attm    = ws + 1840384;       // 64*9*729    = 419904
  float* pscores = ws + 2260288;       // 64*9        = 576
  float* tokens  = ws + 2260864;       // 64*4*1152   = 294912
  float* gsig    = ws + 2555776;       // 64*4        = 256
  __bf16* Bpk    = (__bf16*)(ws + 2556032);  // 5*36*16384 bf16 = 5.9MB

  float* out       = (float*)d_out;
  float* out_major = out;              // 64*4*1152 = 294912
  float* out_final = out + 294912;     // 8*1152    = 9216
  float* out_preds = out + 304128;     // 8*9       = 72

  pack_w_kernel<<<dim3(KSTEPS, 5), 256, 0, stream>>>(sig_w, k_w, Bpk);
  pooled_kernel<<<B_, 256, 0, stream>>>(ov, pool_w, pool_b, pooled);
  cls_kernel<<<B_, 256, 0, stream>>>(pooled, cls_w1, cls_b1, cls_w2, cls_b2, cls);
  prompt_kernel<<<NL_ * B_ * NE_, 256, 0, stream>>>(emo, cls, prompt_w, prompt_b, prompts);
  sig_scores_kernel<<<dim3(46, T_, NL_), 256, 0, stream>>>(sig, Bpk, sig_b, prompts, scores);
  softmax_kernel<<<dim3(NE_, T_), 256, 0, stream>>>(scores, attm, pscores);
  tokens_kernel<<<T_, 256, 0, stream>>>(attm, pscores, ov, tokens, gsig);
  attn_kernel<<<T_, 256, 0, stream>>>(ov, tokens, q_w, q_b,
                                      Bpk + (size_t)NL_ * KSTEPS * BTILE, k_b, out_major);
  final_kernel<<<B_, 256, 0, stream>>>(out_major, gsig, pred_w1, pred_b1, pred_w2, pred_b2,
                                       out_final, out_preds);
}
